// xLSTMInference_70428873719989
// MI455X (gfx1250) — compile-verified, matrix operand order checked by reading, not run on hardware
//
// xLSTM (sLSTM) forward inference for MI455X / gfx1250 (CDNA5, wave32, WMMA).
//
// Phase 1: persistent-grid recurrent scan. Wx+R = 134 MB fp32 < 192 MB L2, so
//          weights stay L2-resident across all 256 steps. bf16 WMMA 16x16x32
//          with B padded 4 -> 16; fp32 gate math; device-scope atomic barrier
//          once per layer-step (1024 total).
// Phase 2: LM head GEMM [1024 x 1024] x [1024 x 50257]; HBM-bound (~412 MB at
//          23.3 TB/s ~= 18 us). head_w tiles staged fp32 -> LDS with the
//          Tensor Data Mover (tensor_load_to_lds + s_wait_tensorcnt), double
//          buffered so the DMA overlaps WMMA compute.

#include <hip/hip_runtime.h>
#include <hip/hip_bf16.h>

#define VV 50257
#define DD 1024
#define LL 4
#define BB 4
#define SS 256
#define NPAD 16
#define NWG 32   // persistent workgroups in the scan kernel

typedef __attribute__((ext_vector_type(16))) __bf16 v16bf;
typedef __attribute__((ext_vector_type(8)))  __bf16 v8bf;
typedef __attribute__((ext_vector_type(8)))  float  v8f;
typedef __attribute__((ext_vector_type(4)))  unsigned int u32x4;
typedef __attribute__((ext_vector_type(8)))  int    i32x8;
typedef __attribute__((ext_vector_type(4)))  int    i32x4;

// ---------------- ws layout (bytes) ----------------
#define XBUF_BYTES   ((size_t)SS * NPAD * DD * 2)           // 8 MiB  bf16 [t][nb][d]
#define HBUF_BYTES   ((size_t)2 * LL * NPAD * DD * 2)       // 256 KiB bf16 [par][l][nb][d]
#define YS_BYTES     ((size_t)SS * BB * DD * 2)             // 2 MiB  bf16 [t*4+b][d]
#define HBUF_OFF     (XBUF_BYTES)
#define YS_OFF       (XBUF_BYTES + HBUF_BYTES)
#define BAR_OFF      (XBUF_BYTES + HBUF_BYTES + YS_BYTES)

// ---------------- WMMA helpers ----------------
__device__ __forceinline__ v8f wmma_bf16(v16bf a, v16bf b, v8f c) {
  // v_wmma_f32_16x16x32_bf16  D = A(16x32) * B(32x16) + C
  return __builtin_amdgcn_wmma_f32_16x16x32_bf16(false, a, false, b, (short)0, c,
                                                 false, false);
}

// A fragment (16x32, bf16) from fp32 row-major weights, converted in VALU.
// Layout per ISA: lanes 0-15 row M=lane, K = k0+{0..7,16..23};
//                 lanes 16-31 row M=lane-16, K = k0+{8..15,24..31}.
__device__ __forceinline__ v16bf a_frag_from_f32(const float* __restrict__ rowbase,
                                                 int ld, int k0) {
  const int lane = threadIdx.x & 31;
  const float* p = rowbase + (size_t)(lane & 15) * ld + k0 + ((lane & 16) ? 8 : 0);
  const float4* p4 = reinterpret_cast<const float4*>(p);
  float4 q0 = p4[0], q1 = p4[1], q2 = p4[4], q3 = p4[5];
  v16bf a;
  a[0]=(__bf16)q0.x; a[1]=(__bf16)q0.y; a[2]=(__bf16)q0.z; a[3]=(__bf16)q0.w;
  a[4]=(__bf16)q1.x; a[5]=(__bf16)q1.y; a[6]=(__bf16)q1.z; a[7]=(__bf16)q1.w;
  a[8]=(__bf16)q2.x; a[9]=(__bf16)q2.y; a[10]=(__bf16)q2.z; a[11]=(__bf16)q2.w;
  a[12]=(__bf16)q3.x; a[13]=(__bf16)q3.y; a[14]=(__bf16)q3.z; a[15]=(__bf16)q3.w;
  return a;
}

// 16-bit fragment from bf16 memory with the same striping (used for A from ys
// and for B from n-major activation buffers: lane -> N column, runs along K).
__device__ __forceinline__ v16bf frag_from_bf16(const __bf16* __restrict__ base,
                                                int ld, int k0) {
  const int lane = threadIdx.x & 31;
  const __bf16* p = base + (size_t)(lane & 15) * ld + k0 + ((lane & 16) ? 8 : 0);
  v8bf r0 = *reinterpret_cast<const v8bf*>(p);
  v8bf r1 = *reinterpret_cast<const v8bf*>(p + 16);
  v16bf v;
  #pragma unroll
  for (int j = 0; j < 8; ++j) { v[j] = r0[j]; v[8 + j] = r1[j]; }
  return v;
}

// B fragment (32x16) from a [32][128] fp32 LDS tile filled by the TDM.
__device__ __forceinline__ v16bf b_frag_from_f32lds(const float* base, int c0) {
  const int lane = threadIdx.x & 31;
  const int c = c0 + (lane & 15);
  const float* p = base + (size_t)((lane & 16) ? 8 : 0) * 128 + c;
  v16bf v;
  #pragma unroll
  for (int j = 0; j < 8; ++j) {
    v[j]     = (__bf16)p[(size_t)j * 128];
    v[8 + j] = (__bf16)p[(size_t)(16 + j) * 128];
  }
  return v;
}

// ---------------- device-scope grid barrier (monotonic counter) ----------------
__device__ __forceinline__ void grid_barrier(unsigned* counter, unsigned nwg) {
  __syncthreads();
  if (threadIdx.x == 0) {
    unsigned arrive = __hip_atomic_fetch_add(counter, 1u, __ATOMIC_ACQ_REL,
                                             __HIP_MEMORY_SCOPE_AGENT);
    unsigned target = (arrive / nwg + 1u) * nwg;
    while (__hip_atomic_load(counter, __ATOMIC_ACQUIRE,
                             __HIP_MEMORY_SCOPE_AGENT) < target) {
      __builtin_amdgcn_s_sleep(1);
    }
  }
  __syncthreads();
}

// ---------------- TDM: stage a 32(K) x 128(N) fp32 tile of head_w into LDS ----
__device__ __forceinline__ void tdm_load_tile(const float* hw, int k0, int cb,
                                              float* ldsdst) {
  const unsigned long long ga =
      (unsigned long long)(size_t)(hw + (size_t)k0 * VV + cb);
  const unsigned lds_addr = (unsigned)(size_t)ldsdst;  // flat[31:0] == LDS offset
  const unsigned dim0 = (unsigned)(VV - cb);  // remaining cols (OOB -> zero fill)
  const unsigned dim1 = (unsigned)(DD - k0);  // remaining rows
  u32x4 g0;
  g0[0] = 1u;                                        // count = 1 valid descriptor
  g0[1] = lds_addr;                                  // lds_addr [63:32]
  g0[2] = (unsigned)(ga & 0xFFFFFFFFull);            // global_addr [95:64]
  g0[3] = (unsigned)((ga >> 32) & 0x01FFFFFFull)     // global_addr [120:96]
        | (2u << 30);                                // type = 2 ("image")
  i32x8 g1;
  g1[0] = (int)(2u << 16);                           // data_size = 4 bytes
  g1[1] = (int)((dim0 & 0xFFFFu) << 16);             // tensor_dim0 lo
  g1[2] = (int)(((dim0 >> 16) & 0xFFFFu) | ((dim1 & 0xFFFFu) << 16));
  g1[3] = (int)(((dim1 >> 16) & 0xFFFFu) | (128u << 16));  // tile_dim0 = 128
  g1[4] = (int)32u;                                  // tile_dim1 = 32, tile_dim2 = 0
  g1[5] = (int)(unsigned)VV;                         // tensor_dim0_stride = V
  g1[6] = 0;
  g1[7] = 0;
  i32x4 z4 = {0, 0, 0, 0};
#if __has_builtin(__builtin_amdgcn_tensor_load_to_lds)
# if defined(__clang_major__) && (__clang_major__ >= 23)
  i32x8 z8 = {0, 0, 0, 0, 0, 0, 0, 0};
  __builtin_amdgcn_tensor_load_to_lds(g0, g1, z4, z4, z8, 0);
# else
  __builtin_amdgcn_tensor_load_to_lds(g0, g1, z4, z4, 0);
# endif
#else
  asm volatile("tensor_load_to_lds %0, %1, %2, %3"
               :: "s"(g0), "s"(g1), "s"(z4), "s"(z4) : "memory");
#endif
}

// ---------------- kernel 1: embedding gather + state/pad init ----------------
__global__ void __launch_bounds__(256) embed_init_kernel(
    const int* __restrict__ ids, const float* __restrict__ emb,
    __bf16* __restrict__ xbuf, __bf16* hbuf, unsigned* bar) {
  const size_t i0 = (size_t)blockIdx.x * blockDim.x + threadIdx.x;
  const size_t stride = (size_t)gridDim.x * blockDim.x;
  const size_t NX = (size_t)SS * NPAD * DD;
  for (size_t idx = i0; idx < NX; idx += stride) {
    const size_t d = idx % DD;
    const size_t nb = (idx / DD) % NPAD;
    const size_t t = idx / ((size_t)NPAD * DD);
    __bf16 v = (__bf16)0.0f;
    if (nb < BB) {
      const int id = ids[nb * SS + t];
      v = (__bf16)emb[(size_t)id * DD + d];
    }
    xbuf[idx] = v;
  }
  const size_t NH = (size_t)2 * LL * NPAD * DD;   // zero both parities of h
  for (size_t idx = i0; idx < NH; idx += stride) hbuf[idx] = (__bf16)0.0f;
  if (i0 == 0) *bar = 0u;
}

// ---------------- kernel 2: persistent recurrent scan ----------------
// 32 WGs x 256 threads (8 waves). WG owns 32 features; wave w -> gate w>>1,
// feature half w&1 -> one 16x16 WMMA output tile. h ping-pongs by t-parity.
__global__ void __launch_bounds__(256) xlstm_scan_kernel(
    const float* __restrict__ Wx, const float* __restrict__ R,
    const float* __restrict__ bias, const __bf16* __restrict__ xbuf,
    __bf16* hbuf, __bf16* __restrict__ ysbuf, unsigned* bar) {
  __shared__ float pre_lds[4 * 32 * NPAD];  // [gate][feat 0..31][batch col], 8KB
  const int tid = threadIdx.x;
  const int lane = tid & 31;
  const int w = tid >> 5;
  const int g = w >> 1;        // gate 0..3 (z, i, f, o)
  const int fh = w & 1;        // feature half within the WG's 32 features
  const int f0 = blockIdx.x * 32;
  const int rbase = (lane >> 4) * 8;   // C-layout: M = r + (lane>=16 ? 8 : 0)
  const int col = lane & 15;           // C-layout: N = lane & 15
  const int fl = tid >> 2;             // state-update thread: feature (tid<128)
  const int bb = tid & 3;              //                      batch

  float c_s[LL], n_s[LL], m_s[LL];
  #pragma unroll
  for (int l = 0; l < LL; ++l) { c_s[l] = 0.f; n_s[l] = 0.f; m_s[l] = 0.f; }

  #pragma unroll 1
  for (int t = 0; t < SS; ++t) {
    const int par = t & 1;
    #pragma unroll
    for (int l = 0; l < LL; ++l) {
      const __bf16* inb = (l == 0)
          ? (xbuf + (size_t)t * NPAD * DD)
          : (hbuf + (size_t)(par * LL + (l - 1)) * NPAD * DD);
      const __bf16* hpb = hbuf + (size_t)((par ^ 1) * LL + l) * NPAD * DD;
      const float* Wrow = Wx + (size_t)((l * 4 + g) * DD + f0 + fh * 16) * DD;
      const float* Rrow = R  + (size_t)((l * 4 + g) * DD + f0 + fh * 16) * DD;

      v8f acc = {0.f, 0.f, 0.f, 0.f, 0.f, 0.f, 0.f, 0.f};
      #pragma unroll 2
      for (int k = 0; k < DD; k += 32)
        acc = wmma_bf16(a_frag_from_f32(Wrow, DD, k),
                        frag_from_bf16(inb, DD, k), acc);
      #pragma unroll 2
      for (int k = 0; k < DD; k += 32)
        acc = wmma_bf16(a_frag_from_f32(Rrow, DD, k),
                        frag_from_bf16(hpb, DD, k), acc);

      const float* bp = bias + (size_t)(l * 4 + g) * DD + f0 + fh * 16 + rbase;
      #pragma unroll
      for (int r = 0; r < 8; ++r)
        pre_lds[(g * 32 + fh * 16 + rbase + r) * NPAD + col] = acc[r] + bp[r];
      __syncthreads();

      if (tid < 128) {  // one thread per (feature, batch) pair; state in regs
        const float pz = pre_lds[(0 * 32 + fl) * NPAD + bb];
        const float pi = pre_lds[(1 * 32 + fl) * NPAD + bb];
        const float pf = pre_lds[(2 * 32 + fl) * NPAD + bb];
        const float po = pre_lds[(3 * 32 + fl) * NPAD + bb];
        const float z  = tanhf(pz);
        const float o  = 1.0f / (1.0f + expf(-po));
        const float mn = fmaxf(pf + m_s[l], pi);
        const float ig = expf(pi - mn);
        const float fg = expf(pf + m_s[l] - mn);
        const float cn = fg * c_s[l] + ig * z;
        const float nn = fg * n_s[l] + ig;
        const float h  = o * (cn / fmaxf(nn, 1e-6f));
        c_s[l] = cn; n_s[l] = nn; m_s[l] = mn;
        const __bf16 h16 = (__bf16)h;
        hbuf[((size_t)(par * LL + l) * NPAD + bb) * DD + f0 + fl] = h16;
        if (l == LL - 1)
          ysbuf[((size_t)t * BB + bb) * DD + f0 + fl] = h16;
      }
      grid_barrier(bar, NWG);   // publish h[l] before any WG starts layer l+1
    }
  }
}

// ---------------- kernel 3: LM head GEMM with TDM-staged B tiles ----------------
__global__ void __launch_bounds__(256) head_gemm_kernel(
    const __bf16* __restrict__ ys, const float* __restrict__ hw,
    const float* __restrict__ hb, float* __restrict__ out) {
  __shared__ float ldsB[2][32][128];  // double-buffered fp32 tile, 32 KB
  const int tid = threadIdx.x, lane = tid & 31, w = tid >> 5;
  const int mi = w & 3, ni = w >> 2;           // 8 waves: 4 (M) x 2 (N-half)
  const int m0 = blockIdx.y * 64 + mi * 16;
  const int cb = blockIdx.x * 128;

  v8f zero = {0.f, 0.f, 0.f, 0.f, 0.f, 0.f, 0.f, 0.f};
  v8f acc[4] = {zero, zero, zero, zero};

  if (w == 0) tdm_load_tile(hw, 0, cb, &ldsB[0][0][0]);  // prologue DMA
  #pragma unroll 1
  for (int k = 0; k < DD; k += 32) {
    const int buf = (k >> 5) & 1;
    if (w == 0) __builtin_amdgcn_s_wait_tensorcnt(0);  // current tile landed
    __syncthreads();                                   // ...and prev reads done
    if (w == 0 && (k + 32) < DD)
      tdm_load_tile(hw, k + 32, cb, &ldsB[buf ^ 1][0][0]);  // overlap next DMA
    v16bf a = frag_from_bf16(ys + (size_t)m0 * DD, DD, k);
    #pragma unroll
    for (int nt = 0; nt < 4; ++nt) {
      v16bf b = b_frag_from_f32lds(&ldsB[buf][0][0], ni * 64 + nt * 16);
      acc[nt] = wmma_bf16(a, b, acc[nt]);
    }
  }

  const int col = lane & 15, rbase = (lane >> 4) * 8;
  #pragma unroll
  for (int nt = 0; nt < 4; ++nt) {
    const int v = cb + ni * 64 + nt * 16 + col;
    if (v < VV) {
      const float bv = hb[v];
      #pragma unroll
      for (int r = 0; r < 8; ++r) {
        const int m = m0 + rbase + r;       // m = t*4 + b
        const int tt = m >> 2, b_ = m & 3;
        out[((size_t)b_ * SS + tt) * (size_t)VV + v] = acc[nt][r] + bv;
      }
    }
  }
}

// ---------------- launch ----------------
extern "C" void kernel_launch(void* const* d_in, const int* in_sizes, int n_in,
                              void* d_out, int out_size, void* d_ws, size_t ws_size,
                              hipStream_t stream) {
  (void)in_sizes; (void)n_in; (void)out_size; (void)ws_size;
  const int*   ids  = (const int*)d_in[0];
  const float* emb  = (const float*)d_in[1];
  const float* Wx   = (const float*)d_in[2];
  const float* R    = (const float*)d_in[3];
  const float* bias = (const float*)d_in[4];
  const float* hw   = (const float*)d_in[5];
  const float* hb   = (const float*)d_in[6];
  float* out = (float*)d_out;
  char* ws = (char*)d_ws;
  __bf16*   xbuf = (__bf16*)(ws);
  __bf16*   hbuf = (__bf16*)(ws + HBUF_OFF);
  __bf16*   ysb  = (__bf16*)(ws + YS_OFF);
  unsigned* bar  = (unsigned*)(ws + BAR_OFF);

  embed_init_kernel<<<dim3(2048), dim3(256), 0, stream>>>(ids, emb, xbuf, hbuf, bar);
  xlstm_scan_kernel<<<dim3(NWG), dim3(256), 0, stream>>>(Wx, R, bias, xbuf, hbuf,
                                                         ysb, bar);
  head_gemm_kernel<<<dim3((VV + 127) / 128, (SS * BB) / 64), dim3(256), 0, stream>>>(
      ysb, hw, hb, out);
}
